// VectorQuantize_65876208386260
// MI455X (gfx1250) — compile-verified, matrix operand order checked by reading, not run on hardware
//
#include <hip/hip_runtime.h>

// ---------------------------------------------------------------------------
// VQ-VAE vector quantize for MI455X (gfx1250, wave32, WMMA).
// Distance GEMM in bf16 via v_wmma_f32_16x16x32_bf16 (fp32 accum).
// Block = 32 token rows; each wave keeps TWO C accumulators sharing every
// B tile (halves L2 traffic, two independent WMMA chains for the XDL pipe).
// ---------------------------------------------------------------------------

typedef __attribute__((ext_vector_type(16))) __bf16        v16bf;
typedef __attribute__((ext_vector_type(8)))  float         v8f;
typedef __attribute__((ext_vector_type(4)))  unsigned int  v4u;

#define DECAY  0.8f
#define OMD    0.2f          /* 1 - DECAY */
#define EPSV   1e-5f

constexpr int Dd = 512;
constexpr int Kk = 8192;
constexpr int Nn = 32768;    // B*T tokens
constexpr int DK = Dd * Kk;  // 4194304
constexpr int ROWS = 32;     // token rows per block (2 M-tiles)

// d_out layout (floats), reference tuple order:
constexpr long long Q_OFF   = 0;                     // quantize_st  [16,2048,512]
constexpr long long IND_OFF = 16777216;              // embed_ind    [16,2048]
constexpr long long LOSS_OFF = IND_OFF + 32768;      // scalar
constexpr long long NE_OFF  = LOSS_OFF + 1;          // new_embed    [512,8192]
constexpr long long NCS_OFF = NE_OFF + DK;           // new_cluster  [8192]
constexpr long long NEA_OFF = NCS_OFF + Kk;          // new_embed_avg[512,8192]

// workspace layout (bytes)
constexpr size_t BPACK_BYTES = (size_t)DK * 2;       // 8.39 MB bf16 codebook, WMMA-B order
constexpr size_t NORM_OFF_B  = BPACK_BYTES;          // K floats
constexpr size_t NSUM_OFF_B  = NORM_OFF_B + (size_t)Kk * 4;

union V16 {
    v4u   q[2];
    v16bf v;
};

static __device__ __forceinline__ unsigned short f2bf(float f) {
    unsigned int u = __builtin_bit_cast(unsigned int, f);
    u += 0x7FFFu + ((u >> 16) & 1u);   // round-to-nearest-even
    return (unsigned short)(u >> 16);
}

// ---------------------------------------------------------------------------
// 1) init: seed EMA buffers with DECAY*old, zero accumulators
// ---------------------------------------------------------------------------
__global__ void vq_init(const float* __restrict__ cluster_size,
                        const float* __restrict__ embed_avg,
                        float* __restrict__ out, float* __restrict__ nsum) {
    int i = blockIdx.x * 256 + threadIdx.x;
    out[NEA_OFF + i] = DECAY * embed_avg[i];
    if (i < Kk) out[NCS_OFF + i] = DECAY * cluster_size[i];
    if (i == 0) { out[LOSS_OFF] = 0.0f; *nsum = 0.0f; }
}

// ---------------------------------------------------------------------------
// 2) per-code squared norms ||e_k||^2
// ---------------------------------------------------------------------------
__global__ void vq_norms(const float* __restrict__ embed, float* __restrict__ norms) {
    int k = blockIdx.x * 256 + threadIdx.x;
    float s = 0.0f;
    for (int d = 0; d < Dd; ++d) {
        float v = embed[(size_t)d * Kk + k];
        s = fmaf(v, v, s);
    }
    norms[k] = s;
}

// ---------------------------------------------------------------------------
// 3) pack embed -> bf16 in exact wave32 WMMA B-operand order.
//    B tile (kt, dt) covers cols k = kt*16+n, reduction d = dt*32 + kr.
//    Lane l: n = l&15, h = l>>4.  dword j<4: kr pair = 8h+2j, 8h+2j+1;
//    j>=4: kr pair = 16+8h+2(j-4), +1.  16 bf16 contiguous per lane.
// ---------------------------------------------------------------------------
__global__ void vq_bpack(const float* __restrict__ embed, unsigned short* __restrict__ bp) {
    int i  = blockIdx.x * 256 + threadIdx.x;       // one bf16 element
    int p  = i & 1;
    int j  = (i >> 1) & 7;
    int l  = (i >> 4) & 31;
    int dt = (i >> 9) & 15;
    int kt = i >> 13;
    int n  = l & 15, h = l >> 4;
    int kr = (j < 4) ? (8 * h + 2 * j + p) : (16 + 8 * h + 2 * (j - 4) + p);
    int d  = dt * 32 + kr;
    int k  = kt * 16 + n;
    bp[i] = f2bf(embed[(size_t)d * Kk + k]);
}

// ---------------------------------------------------------------------------
// 4) main: distance GEMM (WMMA bf16) + fused argmin + gather + EMA stats
//    block = 32 token rows; 8 waves each scan 1024 codes with 2 M-tiles.
// ---------------------------------------------------------------------------
__global__ __launch_bounds__(256) void vq_main(const float* __restrict__ x,
                                               const float* __restrict__ embed,
                                               const unsigned short* __restrict__ bp,
                                               const float* __restrict__ norms,
                                               float* __restrict__ out) {
    __shared__ unsigned short sA[ROWS * Dd];   // 32 KB, bf16 bits, row-major [32][512]
    __shared__ float red_v[8][ROWS];
    __shared__ int   red_k[8][ROWS];
    __shared__ int   final_k[ROWS];
    __shared__ float lbuf[256];

    const int tid  = threadIdx.x;
    const int blk  = blockIdx.x;
    const int row0 = blk * ROWS;

    // stage + convert A block: 32 rows x 512 fp32 -> bf16 LDS
    for (int idx = tid; idx < ROWS * Dd; idx += 256)
        sA[idx] = f2bf(x[(size_t)row0 * Dd + idx]);
    __syncthreads();

    const int wave = tid >> 5;
    const int lane = tid & 31;
    const int n    = lane & 15;
    const int h    = lane >> 4;
    const int m    = lane & 15;   // A row within a 16-row tile (both halves)

    float bestv0[8], bestv1[8];
    int   bestk0[8], bestk1[8];
#pragma unroll
    for (int i = 0; i < 8; ++i) {
        bestv0[i] = 3.4e38f; bestk0[i] = 0;
        bestv1[i] = 3.4e38f; bestk1[i] = 0;
    }

    // A operand per tile: vgpr0-3 = d0+8h .. +7, vgpr4-7 = d0+16+8h .. +7
    const unsigned short* a0p0 = &sA[m * Dd + 8 * h];
    const unsigned short* a0p1 = &sA[m * Dd + 16 + 8 * h];
    const unsigned short* a1p0 = &sA[(m + 16) * Dd + 8 * h];
    const unsigned short* a1p1 = &sA[(m + 16) * Dd + 16 + 8 * h];

    const int kt_lo = wave * 64, kt_hi = kt_lo + 64;   // 8 waves x 64 tiles = K/16
    for (int kt = kt_lo; kt < kt_hi; ++kt) {
        v8f c0 = {};
        v8f c1 = {};
        // lane's 16 bf16 per dt step are contiguous; stride 32 lanes * 16 bf16
        const v4u* bptr = (const v4u*)bp + ((size_t)kt * 16 * 32 + lane) * 2;
#pragma unroll 4
        for (int dt = 0; dt < 16; ++dt) {
            V16 a0, a1, b;
            b.q[0]  = bptr[0];                           // global_load_b128 (L2)
            b.q[1]  = bptr[1];
            a0.q[0] = *(const v4u*)(a0p0 + dt * 32);     // ds_load_b128
            a0.q[1] = *(const v4u*)(a0p1 + dt * 32);
            a1.q[0] = *(const v4u*)(a1p0 + dt * 32);
            a1.q[1] = *(const v4u*)(a1p1 + dt * 32);
            bptr   += 64;                                // next dt tile
            c0 = __builtin_amdgcn_wmma_f32_16x16x32_bf16(
                     false, a0.v, false, b.v, (short)0, c0, false, false);
            c1 = __builtin_amdgcn_wmma_f32_16x16x32_bf16(
                     false, a1.v, false, b.v, (short)0, c1, false, false);
        }
        // dist = ||e||^2 - 2 f.e  (||f||^2 constant per row, irrelevant to argmin)
        const float nk   = norms[kt * 16 + n];
        const int   kidx = kt * 16 + n;
#pragma unroll
        for (int i = 0; i < 8; ++i) {
            float d0 = fmaf(-2.0f, c0[i], nk);
            float d1 = fmaf(-2.0f, c1[i], nk);
            if (d0 < bestv0[i]) { bestv0[i] = d0; bestk0[i] = kidx; }  // '<' keeps lowest k
            if (d1 < bestv1[i]) { bestv1[i] = d1; bestk1[i] = kidx; }
        }
    }

    // argmin butterfly inside each 16-lane group (stays within half-wave)
#pragma unroll
    for (int off = 1; off < 16; off <<= 1) {
#pragma unroll
        for (int i = 0; i < 8; ++i) {
            float ov; int ok;
            ov = __shfl_xor(bestv0[i], off, 32);
            ok = __shfl_xor(bestk0[i], off, 32);
            if (ov < bestv0[i] || (ov == bestv0[i] && ok < bestk0[i])) {
                bestv0[i] = ov; bestk0[i] = ok;
            }
            ov = __shfl_xor(bestv1[i], off, 32);
            ok = __shfl_xor(bestk1[i], off, 32);
            if (ov < bestv1[i] || (ov == bestv1[i] && ok < bestk1[i])) {
                bestv1[i] = ov; bestk1[i] = ok;
            }
        }
    }
    if ((lane & 15) == 0) {        // lane 0 -> rows i+0/8, lane 16 -> rows i+8 halves
#pragma unroll
        for (int i = 0; i < 8; ++i) {
            red_v[wave][i + 8 * h]      = bestv0[i];
            red_k[wave][i + 8 * h]      = bestk0[i];
            red_v[wave][16 + i + 8 * h] = bestv1[i];
            red_k[wave][16 + i + 8 * h] = bestk1[i];
        }
    }
    __syncthreads();

    if (tid < ROWS) {               // combine the 8 waves per row
        float bv = red_v[0][tid];
        int   bk = red_k[0][tid];
        for (int w = 1; w < 8; ++w) {
            float ov = red_v[w][tid]; int ok = red_k[w][tid];
            if (ov < bv || (ov == bv && ok < bk)) { bv = ov; bk = ok; }
        }
        final_k[tid] = bk;
        out[IND_OFF + row0 + tid] = (float)bk;
        atomicAdd(&out[NCS_OFF + bk], OMD);              // EMA one-hot count
    }
    __syncthreads();

    // gather quantize_st, commitment loss, EMA embed-sum scatter
    float ls = 0.0f;
    for (int r = 0; r < ROWS; ++r) {
        const int k = final_k[r];
        for (int d = tid; d < Dd; d += 256) {
            float q  = embed[(size_t)d * Kk + k];
            float xv = x[((size_t)(row0 + r)) * Dd + d];
            out[Q_OFF + ((size_t)(row0 + r)) * Dd + d] = q;   // x + sg(q-x) == q
            float df = q - xv;
            ls = fmaf(df, df, ls);
            atomicAdd(&out[NEA_OFF + (size_t)d * Kk + k], OMD * xv);
        }
    }
    lbuf[tid] = ls;
    __syncthreads();
    for (int s = 128; s > 0; s >>= 1) {
        if (tid < s) lbuf[tid] += lbuf[tid + s];
        __syncthreads();
    }
    if (tid == 0)
        atomicAdd(&out[LOSS_OFF], lbuf[0] * (1.0f / ((float)Nn * (float)Dd)));
}

// ---------------------------------------------------------------------------
// 5) n = sum(new_cluster_size)
// ---------------------------------------------------------------------------
__global__ void vq_nsum(const float* __restrict__ ncs, float* __restrict__ nsum) {
    __shared__ float buf[256];
    int i = blockIdx.x * 256 + threadIdx.x;
    buf[threadIdx.x] = ncs[i];
    __syncthreads();
    for (int s = 128; s > 0; s >>= 1) {
        if (threadIdx.x < s) buf[threadIdx.x] += buf[threadIdx.x + s];
        __syncthreads();
    }
    if (threadIdx.x == 0) atomicAdd(nsum, buf[0]);
}

// ---------------------------------------------------------------------------
// 6) new_embed = new_embed_avg / cs(k)
// ---------------------------------------------------------------------------
__global__ void vq_final(const float* __restrict__ nsum, float* __restrict__ out) {
    int i = blockIdx.x * 256 + threadIdx.x;
    int k = i & (Kk - 1);
    float nn  = *nsum;
    float ncs = out[NCS_OFF + k];
    float cs  = (ncs + EPSV) / (nn + (float)Kk * EPSV) * nn;
    out[NE_OFF + i] = out[NEA_OFF + i] / cs;
}

// ---------------------------------------------------------------------------
extern "C" void kernel_launch(void* const* d_in, const int* in_sizes, int n_in,
                              void* d_out, int out_size, void* d_ws, size_t ws_size,
                              hipStream_t stream) {
    const float* x            = (const float*)d_in[0];
    const float* embed        = (const float*)d_in[1];
    const float* cluster_size = (const float*)d_in[2];
    const float* embed_avg    = (const float*)d_in[3];
    float* out = (float*)d_out;

    char* wsb = (char*)d_ws;
    unsigned short* bpack = (unsigned short*)wsb;
    float* norms = (float*)(wsb + NORM_OFF_B);
    float* nsum  = (float*)(wsb + NSUM_OFF_B);

    vq_init <<<DK / 256, 256, 0, stream>>>(cluster_size, embed_avg, out, nsum);
    vq_norms<<<Kk / 256, 256, 0, stream>>>(embed, norms);
    vq_bpack<<<DK / 256, 256, 0, stream>>>(embed, bpack);
    vq_main <<<Nn / ROWS, 256, 0, stream>>>(x, embed, bpack, norms, out);
    vq_nsum <<<Kk / 256, 256, 0, stream>>>(out + NCS_OFF, nsum);
    vq_final<<<DK / 256, 256, 0, stream>>>(nsum, out);
}
